// TMDLayer_75462575391148
// MI455X (gfx1250) — matrix operand order repaired and checked by reading, not compiled, not run on hardware
//
#include <hip/hip_runtime.h>

#define Nn 4096
#define Cc 512
#define Ll 64
#define EPSV 0.25f

typedef __attribute__((ext_vector_type(16))) __bf16 v16bf;
typedef __attribute__((ext_vector_type(8)))  float  v8f;

struct __align__(16) U4 { unsigned int x, y, z, w; };
struct __align__(16) F4 { float x, y, z, w; };

union Frag { unsigned int u[8]; U4 q[2]; v16bf v; };

// gfx1250 async global->LDS copy path (ASYNCcnt-tracked), guarded so absence
// of the builtin degrades to plain copies instead of breaking the build.
#if defined(__HIP_DEVICE_COMPILE__) && __has_builtin(__builtin_amdgcn_global_load_async_to_lds_b128)
#define HAVE_ASYNC 1
#else
#define HAVE_ASYNC 0
#endif

#if HAVE_ASYNC
typedef int v4i __attribute__((vector_size(16)));
typedef __attribute__((address_space(1))) v4i* gas_v4i;
typedef __attribute__((address_space(3))) v4i* las_v4i;
#endif

__device__ __forceinline__ void cp16(const void* g, void* l) {
#if HAVE_ASYNC
    __builtin_amdgcn_global_load_async_to_lds_b128(
        (gas_v4i)(v4i*)g, (las_v4i)(v4i*)l, 0, 0);
#else
    *(U4*)l = *(const U4*)g;
#endif
}

__device__ __forceinline__ void async_wait_all() {
#if HAVE_ASYNC
#if __has_builtin(__builtin_amdgcn_s_wait_asynccnt)
    __builtin_amdgcn_s_wait_asynccnt(0);
#else
    asm volatile("s_wait_asynccnt 0" ::: "memory");
#endif
#endif
}

__device__ __forceinline__ unsigned short f2bf(float f) {
    unsigned int u = __float_as_uint(f);
    u += 0x7FFFu + ((u >> 16) & 1u);          // round-to-nearest-even
    return (unsigned short)(u >> 16);
}

// ---------------------------------------------------------------- conversions
__global__ void cvt_bf16_kernel(const float* __restrict__ src,
                                unsigned short* __restrict__ dst, int n) {
    int n4 = n >> 2;
    int stride = gridDim.x * blockDim.x;
    for (int i = blockIdx.x * blockDim.x + threadIdx.x; i < n4; i += stride) {
        F4 v = *(const F4*)(src + (size_t)i * 4);
        unsigned int lo = (unsigned int)f2bf(v.x) | ((unsigned int)f2bf(v.y) << 16);
        unsigned int hi = (unsigned int)f2bf(v.z) | ((unsigned int)f2bf(v.w) << 16);
        unsigned int* d32 = (unsigned int*)(dst + (size_t)i * 4);
        d32[0] = lo; d32[1] = hi;
    }
}

// pack rows (2k,2k+1) of an f32 [K][Nd] matrix into bf16-pair dwords [K/2][Nd]
__global__ void cvt_pack_kernel(const float* __restrict__ src,
                                unsigned int* __restrict__ dst, int K, int Nd) {
    int total = (K >> 1) * Nd;
    int stride = gridDim.x * blockDim.x;
    for (int i = blockIdx.x * blockDim.x + threadIdx.x; i < total; i += stride) {
        int kp = i / Nd, n = i - kp * Nd;
        float lo = src[(size_t)(2 * kp) * Nd + n];
        float hi = src[(size_t)(2 * kp + 1) * Nd + n];
        dst[i] = (unsigned int)f2bf(lo) | ((unsigned int)f2bf(hi) << 16);
    }
}

// ---------------------------------------------------------------- pi head
__global__ void pi_kernel(const float* __restrict__ x,  const float* __restrict__ Wp,
                          const float* __restrict__ bp, const float* __restrict__ W1,
                          const float* __restrict__ b1, const float* __restrict__ W2,
                          const float* __restrict__ b2, float* __restrict__ pi) {
    __shared__ float z[Ll];
    __shared__ float red[Ll];
    int n = blockIdx.x, t = threadIdx.x;          // 64 threads
    const float* xr = x + (size_t)n * Cc;
    float a = bp[t];
    for (int c = 0; c < Cc; ++c) a += xr[c] * Wp[c * Ll + t];
    z[t] = a;
    __syncthreads();
    float p = 0.f;
    for (int j = 0; j < Cc / Ll; ++j) {
        int c = t + j * Ll;
        float h = b1[c];
        #pragma unroll 8
        for (int k = 0; k < Ll; ++k) h += z[k] * W1[k * Cc + c];
        h = fmaxf(h, 0.f);
        p += h * W2[c];
    }
    red[t] = p; __syncthreads();
    for (int sft = 32; sft > 0; sft >>= 1) {
        if (t < sft) red[t] += red[t + sft];
        __syncthreads();
    }
    if (t == 0) pi[n] = 1.f / (1.f + __expf(-(red[0] + b2[0])));
}

// ---------------------------------------------------------------- q rowsum -> s = pi/q
__global__ void rowsum_s_kernel(const float* __restrict__ adj,
                                const float* __restrict__ pi, float* __restrict__ s) {
    __shared__ float red[256];
    int i = blockIdx.x, t = threadIdx.x;
    const F4* row4 = (const F4*)(adj + (size_t)i * Nn);
    float acc = 0.f;
    for (int j = t; j < Nn / 4; j += 256) { F4 v = row4[j]; acc += v.x + v.y + v.z + v.w; }
    red[t] = acc; __syncthreads();
    for (int sft = 128; sft > 0; sft >>= 1) {
        if (t < sft) red[t] += red[t + sft];
        __syncthreads();
    }
    if (t == 0) s[i] = pi[i] / red[0];
}

// ---------------------------------------------------------------- d = adj@s + 1e-5 ; rs = dt/(eps*d)
__global__ void d_rs_kernel(const float* __restrict__ adj, const float* __restrict__ s,
                            const float* __restrict__ dtp, float* __restrict__ rs) {
    __shared__ float red[256];
    int i = blockIdx.x, t = threadIdx.x;
    const F4* row4 = (const F4*)(adj + (size_t)i * Nn);
    const F4* s4 = (const F4*)s;
    float acc = 0.f;
    for (int j = t; j < Nn / 4; j += 256) {
        F4 a = row4[j]; F4 b = s4[j];
        acc += a.x * b.x + a.y * b.y + a.z * b.z + a.w * b.w;
    }
    red[t] = acc; __syncthreads();
    for (int sft = 128; sft > 0; sft >>= 1) {
        if (t < sft) red[t] += red[t + sft];
        __syncthreads();
    }
    if (t == 0) { float d = red[0] + 1e-5f; rs[i] = dtp[0] / (EPSV * d); }
}

// ---------------------------------------------------------------- WMMA GEMM core
// Block tile 128x128x32, 8 waves (2x4), wave tile 64x32 = 4x2 WMMA 16x16 tiles.
// A row-major [.,K] bf16; B pre-pair-packed dwords [K/2][ldbDW]:
//   Bpk[k/2][n] = bf16(B[k][n]) | bf16(B[k+1][n])<<16   (exact WMMA B layout).
#define SMA_ELEMS (128 * 40)
#define SMB_DW    (16 * 132)

__device__ __forceinline__ void stage_tiles(const unsigned short* __restrict__ A,
                                            const unsigned int* __restrict__ Bpk,
                                            int K, int ldbDW, int m0, int n0, int k0,
                                            unsigned short* smA, unsigned int* smB, int tid) {
    int sr = tid >> 2, sc = tid & 3;           // A: row (0..63) + 64, 8-elem chunk
    cp16(A + (size_t)(m0 + sr) * K + k0 + sc * 8,       smA + sr * 40 + sc * 8);
    cp16(A + (size_t)(m0 + sr + 64) * K + k0 + sc * 8,  smA + (sr + 64) * 40 + sc * 8);
    int kk = tid >> 4, nb = (tid & 15) * 8;    // B: K-pair row (0..15), col group
    const unsigned int* gB = Bpk + (size_t)((k0 >> 1) + kk) * ldbDW + n0 + nb;
    unsigned int* lB = smB + kk * 132 + nb;
    cp16(gB, lB);
    cp16(gB + 4, lB + 4);
}

__device__ __forceinline__ void gemm_tiles(const unsigned short* __restrict__ A,
                                           const unsigned int* __restrict__ Bpk,
                                           int K, int ldbDW, int m0, int n0,
                                           unsigned short* smA, unsigned int* smB,
                                           v8f (&acc)[4][2]) {
    int tid = threadIdx.x;
    int lane = tid & 31, wave = tid >> 5;
    int wr = wave >> 2, wc = wave & 3;
    int lrow = lane & 15, hh = lane >> 4;

    stage_tiles(A, Bpk, K, ldbDW, m0, n0, 0, smA, smB, tid);
    int buf = 0;
    for (int k0 = 0; k0 < K; k0 += 32) {
        unsigned short* curA = smA + buf * SMA_ELEMS;
        unsigned int*   curB = smB + buf * SMB_DW;
        async_wait_all();
        __syncthreads();
        if (k0 + 32 < K)
            stage_tiles(A, Bpk, K, ldbDW, m0, n0, k0 + 32,
                        smA + (buf ^ 1) * SMA_ELEMS, smB + (buf ^ 1) * SMB_DW, tid);

        Frag a[4], b[2];
        #pragma unroll
        for (int mt = 0; mt < 4; ++mt) {
            // A layout (matches B): lanes 0-15 -> K{0..15}, lanes 16-31 -> K{16..31}
            const unsigned short* base = curA + (wr * 64 + mt * 16 + lrow) * 40 + hh * 16;
            a[mt].q[0] = *(const U4*)(base);
            a[mt].q[1] = *(const U4*)(base + 8);
        }
        #pragma unroll
        for (int nt = 0; nt < 2; ++nt) {
            // B k order used here (contiguous, same as A): VGPR v = K{2v,2v+1} (lanes<16) / K{16+2v,17+2v}, N=lane&15
            const unsigned int* bp = curB + hh * 8 * 132 + wc * 32 + nt * 16 + lrow;
            #pragma unroll
            for (int v = 0; v < 8; ++v) b[nt].u[v] = bp[v * 132];
        }
        #pragma unroll
        for (int mt = 0; mt < 4; ++mt)
            #pragma unroll
            for (int nt = 0; nt < 2; ++nt)
                acc[mt][nt] = __builtin_amdgcn_wmma_f32_16x16x32_bf16(
                    false, a[mt].v, false, b[nt].v, (short)0, acc[mt][nt], false, false);
        buf ^= 1;
    }
}

// ---------------------------------------------------------------- target = relu(x@W_f + b_f); Bpk = pack(bf16(s[row]*target))
__global__ __launch_bounds__(256) void gemm_target_kernel(
        const unsigned short* __restrict__ A,   // xbf   [Nn][Cc]
        const unsigned int* __restrict__ Bw,    // wfpk  [Cc/2][Cc] dwords
        const float* __restrict__ bias, const float* __restrict__ s,
        float* __restrict__ target, unsigned int* __restrict__ Bpk) {
    __shared__ __align__(16) unsigned short smA[2 * SMA_ELEMS];
    __shared__ __align__(16) unsigned int  smB[2 * SMB_DW];
    v8f acc[4][2];
    v8f zero = {0.f, 0.f, 0.f, 0.f, 0.f, 0.f, 0.f, 0.f};
    #pragma unroll
    for (int i = 0; i < 4; ++i)
        #pragma unroll
        for (int j = 0; j < 2; ++j) acc[i][j] = zero;

    int m0 = blockIdx.x * 128, n0 = blockIdx.y * 128;
    gemm_tiles(A, Bw, Cc, Cc, m0, n0, smA, smB, acc);

    int tid = threadIdx.x, lane = tid & 31, wave = tid >> 5;
    int wr = wave >> 2, wc = wave & 3, lrow = lane & 15, hh = lane >> 4;
    #pragma unroll
    for (int mt = 0; mt < 4; ++mt)
        #pragma unroll
        for (int nt = 0; nt < 2; ++nt) {
            int rbase = m0 + wr * 64 + mt * 16 + hh * 8;  // C/D layout: M = e + 8*(lane>=16)
            int col = n0 + wc * 32 + nt * 16 + lrow;
            float v[8];
            #pragma unroll
            for (int e = 0; e < 8; ++e) {
                float t = acc[mt][nt][e] + bias[col];
                t = fmaxf(t, 0.0f);
                v[e] = t;
                target[(size_t)(rbase + e) * Cc + col] = t;
            }
            #pragma unroll
            for (int e = 0; e < 8; e += 2) {     // rows rbase+e, rbase+e+1 are a K-pair
                int row = rbase + e;
                unsigned int lo = f2bf(s[row] * v[e]);
                unsigned int hi = f2bf(s[row + 1] * v[e + 1]);
                Bpk[(size_t)(row >> 1) * Cc + col] = lo | (hi << 16);
            }
        }
}

// ---------------------------------------------------------------- out = (1-dt)*target + rs[row]*(adj @ B)
__global__ __launch_bounds__(256) void gemm_big_kernel(
        const unsigned short* __restrict__ adjbf,  // [Nn][Nn] bf16
        const unsigned int* __restrict__ Bpk,      // [Nn/2][Cc] dwords
        const float* __restrict__ target, const float* __restrict__ rs,
        const float* __restrict__ dtp, float* __restrict__ out) {
    __shared__ __align__(16) unsigned short smA[2 * SMA_ELEMS];
    __shared__ __align__(16) unsigned int  smB[2 * SMB_DW];
    v8f acc[4][2];
    v8f zero = {0.f, 0.f, 0.f, 0.f, 0.f, 0.f, 0.f, 0.f};
    #pragma unroll
    for (int i = 0; i < 4; ++i)
        #pragma unroll
        for (int j = 0; j < 2; ++j) acc[i][j] = zero;

    int m0 = blockIdx.x * 128, n0 = blockIdx.y * 128;
    gemm_tiles(adjbf, Bpk, Nn, Cc, m0, n0, smA, smB, acc);

    float dt0 = dtp[0];
    int tid = threadIdx.x, lane = tid & 31, wave = tid >> 5;
    int wr = wave >> 2, wc = wave & 3, lrow = lane & 15, hh = lane >> 4;
    #pragma unroll
    for (int mt = 0; mt < 4; ++mt)
        #pragma unroll
        for (int nt = 0; nt < 2; ++nt)
            #pragma unroll
            for (int e = 0; e < 8; ++e) {
                int row = m0 + wr * 64 + mt * 16 + hh * 8 + e;
                int col = n0 + wc * 32 + nt * 16 + lrow;
                float t = target[(size_t)row * Cc + col];
                out[(size_t)row * Cc + col] = (1.f - dt0) * t + rs[row] * acc[mt][nt][e];
            }
}

// ---------------------------------------------------------------- launch
extern "C" void kernel_launch(void* const* d_in, const int* in_sizes, int n_in,
                              void* d_out, int out_size, void* d_ws, size_t ws_size,
                              hipStream_t stream) {
    const float* x      = (const float*)d_in[0];
    const float* adj    = (const float*)d_in[1];
    const float* W_proj = (const float*)d_in[2];
    const float* b_proj = (const float*)d_in[3];
    const float* W_pi1  = (const float*)d_in[4];
    const float* b_pi1  = (const float*)d_in[5];
    const float* W_pi2  = (const float*)d_in[6];
    const float* b_pi2  = (const float*)d_in[7];
    const float* W_f    = (const float*)d_in[8];
    const float* b_f    = (const float*)d_in[9];
    const float* dt     = (const float*)d_in[10];
    float* out = (float*)d_out;
    (void)in_sizes; (void)n_in; (void)out_size; (void)ws_size;

    char* base = (char*)d_ws; size_t off = 0;
    auto carve = [&](size_t bytes) -> char* {
        char* p = base + off;
        off += (bytes + 255) & ~(size_t)255;
        return p;
    };
    unsigned short* adjbf = (unsigned short*)carve((size_t)Nn * Nn * 2);        // 32 MB
    unsigned short* xbf   = (unsigned short*)carve((size_t)Nn * Cc * 2);        //  4 MB
    unsigned int*   wfpk  = (unsigned int*)carve((size_t)(Cc / 2) * Cc * 4);    // .5 MB
    unsigned int*   Bpk   = (unsigned int*)carve((size_t)(Nn / 2) * Cc * 4);    //  4 MB
    float* pi     = (float*)carve(Nn * 4);
    float* s      = (float*)carve(Nn * 4);
    float* rs     = (float*)carve(Nn * 4);
    float* target = (float*)carve((size_t)Nn * Cc * 4);                          //  8 MB

    cvt_bf16_kernel<<<2048, 256, 0, stream>>>(adj, adjbf, Nn * Nn);
    cvt_bf16_kernel<<<1024, 256, 0, stream>>>(x, xbf, Nn * Cc);
    cvt_pack_kernel<<<512, 256, 0, stream>>>(W_f, wfpk, Cc, Cc);
    pi_kernel<<<Nn, Ll, 0, stream>>>(x, W_proj, b_proj, W_pi1, b_pi1, W_pi2, b_pi2, pi);
    rowsum_s_kernel<<<Nn, 256, 0, stream>>>(adj, pi, s);
    d_rs_kernel<<<Nn, 256, 0, stream>>>(adj, s, dt, rs);
    gemm_target_kernel<<<dim3(Nn / 128, Cc / 128), 256, 0, stream>>>(xbf, wfpk, b_f, s, target, Bpk);
    gemm_big_kernel<<<dim3(Nn / 128, Cc / 128), 256, 0, stream>>>(adjbf, Bpk, target, rs, dt, out);
}
